// Attention_80393197847209
// MI455X (gfx1250) — matrix operand order repaired and checked by reading, not compiled, not run on hardware
//
#include <hip/hip_runtime.h>

// ---------------------------------------------------------------------------
// Types for gfx1250 WMMA (wave32): 16x16x32 f16 -> f32 accumulate
// ---------------------------------------------------------------------------
typedef __attribute__((ext_vector_type(16))) _Float16 h16;
typedef __attribute__((ext_vector_type(8)))  _Float16 h8;
typedef __attribute__((ext_vector_type(8)))  float    f8;

#define NEGBIG (-1e30f)

// ---------------------------------------------------------------------------
// Generic WMMA GEMM:  out[m,n] = alpha * sum_k A[m,k]*W[n,k] (+bias[n]) (+Cadd[m,n])
// A: (M,K) f16 row-major (lda), W: (N,K) f16 row-major (ldw).
// One wave computes a 32(M) x 64(N) strip (2 A-frags x 4 W-frags = 8 WMMAs per
// 32-deep K step, 12 b128 loads); block = 4 waves (128 threads) spanning 256 N.
// Fragment layout used here (contiguous k, same for A and B; not the ISA table order):
//   A (16x32 f16): lane r(0..15) row m; VGPR0-7 hold K = koffA..koffA+15 contiguous,
//                  koffA = (lane>=16)?16:0
//   B (32x16 f16): lane r col n; VGPR0-7 hold K = koffB..koffB+15 contiguous,
//                  koffB = (lane>=16)?16:0
// Requires M % 32 == 0, K % 32 == 0 (true for all uses here).
// ---------------------------------------------------------------------------
__global__ __launch_bounds__(128) void att_gemm_f16_wmma(
    const _Float16* __restrict__ A, const _Float16* __restrict__ W,
    const float* __restrict__ bias, const float* __restrict__ Cadd,
    float* __restrict__ outF, _Float16* __restrict__ outH,
    int M, int N, int K, int lda, int ldw, int ldo,
    long long sA, long long sW, long long sO, long long sC, float alpha)
{
    const int bz   = blockIdx.z;
    const int wave = threadIdx.x >> 5;
    const int lane = threadIdx.x & 31;
    const int m0   = blockIdx.y << 5;
    const int n0   = (blockIdx.x * 4 + wave) << 6;
    if (m0 >= M || n0 >= N) return;

    const int r     = lane & 15;
    const int hi    = lane >> 4;
    const int koffA = hi << 4;   // 0 or 16
    const int koffB = hi << 4;   // 0 or 16

    const _Float16* A0 = A + (long long)bz * sA + (long long)(m0 + r) * lda;
    const _Float16* A1 = A0 + (long long)16 * lda;
    const _Float16* Wb = W + (long long)bz * sW + (long long)(n0 + r) * ldw;
    const _Float16* W0 = Wb;
    const _Float16* W1 = Wb + (long long)16 * ldw;
    const _Float16* W2 = Wb + (long long)32 * ldw;
    const _Float16* W3 = Wb + (long long)48 * ldw;

    union Frag { h16 v; h8 h[2]; };
    f8 acc[2][4];
#pragma unroll
    for (int a = 0; a < 2; ++a)
#pragma unroll
        for (int w = 0; w < 4; ++w) acc[a][w] = (f8){};

    for (int k0 = 0; k0 < K; k0 += 32) {
        Frag fa0, fa1, f0, f1, f2, f3;
        fa0.h[0] = *(const h8*)(A0 + k0 + koffA);
        fa0.h[1] = *(const h8*)(A0 + k0 + koffA + 8);
        fa1.h[0] = *(const h8*)(A1 + k0 + koffA);
        fa1.h[1] = *(const h8*)(A1 + k0 + koffA + 8);
        f0.h[0]  = *(const h8*)(W0 + k0 + koffB);
        f0.h[1]  = *(const h8*)(W0 + k0 + koffB + 8);
        f1.h[0]  = *(const h8*)(W1 + k0 + koffB);
        f1.h[1]  = *(const h8*)(W1 + k0 + koffB + 8);
        f2.h[0]  = *(const h8*)(W2 + k0 + koffB);
        f2.h[1]  = *(const h8*)(W2 + k0 + koffB + 8);
        f3.h[0]  = *(const h8*)(W3 + k0 + koffB);
        f3.h[1]  = *(const h8*)(W3 + k0 + koffB + 8);
        __builtin_prefetch(A0 + k0 + 64, 0, 1);   // global_prefetch
        __builtin_prefetch(W0 + k0 + 64, 0, 1);
        acc[0][0] = __builtin_amdgcn_wmma_f32_16x16x32_f16(false, fa0.v, false, f0.v, (short)0, acc[0][0], false, false);
        acc[0][1] = __builtin_amdgcn_wmma_f32_16x16x32_f16(false, fa0.v, false, f1.v, (short)0, acc[0][1], false, false);
        acc[0][2] = __builtin_amdgcn_wmma_f32_16x16x32_f16(false, fa0.v, false, f2.v, (short)0, acc[0][2], false, false);
        acc[0][3] = __builtin_amdgcn_wmma_f32_16x16x32_f16(false, fa0.v, false, f3.v, (short)0, acc[0][3], false, false);
        acc[1][0] = __builtin_amdgcn_wmma_f32_16x16x32_f16(false, fa1.v, false, f0.v, (short)0, acc[1][0], false, false);
        acc[1][1] = __builtin_amdgcn_wmma_f32_16x16x32_f16(false, fa1.v, false, f1.v, (short)0, acc[1][1], false, false);
        acc[1][2] = __builtin_amdgcn_wmma_f32_16x16x32_f16(false, fa1.v, false, f2.v, (short)0, acc[1][2], false, false);
        acc[1][3] = __builtin_amdgcn_wmma_f32_16x16x32_f16(false, fa1.v, false, f3.v, (short)0, acc[1][3], false, false);
    }

    // C/D layout: element (vgpr v, lane): m = m0 + a*16 + v + 8*hi, n = n0 + nt*16 + (lane&15)
    const long long ob = (long long)bz * sO;
    const long long cb = (long long)bz * sC;
#pragma unroll
    for (int a = 0; a < 2; ++a) {
#pragma unroll
        for (int nt = 0; nt < 4; ++nt) {
            const int n  = n0 + nt * 16 + r;
            const float bv = bias ? bias[n] : 0.0f;
#pragma unroll
            for (int v = 0; v < 8; ++v) {
                const int m = m0 + a * 16 + v + (hi << 3);
                const long long o = (long long)m * ldo + n;
                float val = alpha * acc[a][nt][v] + bv;
                if (Cadd) val += Cadd[cb + o];
                if (outF) outF[ob + o] = val;
                else      outH[ob + o] = (_Float16)val;
            }
        }
    }
}

// ---------------------------------------------------------------------------
// Elementwise helpers
// ---------------------------------------------------------------------------
__global__ void att_f32_to_f16(const float* __restrict__ in, _Float16* __restrict__ out, long long n)
{
    for (long long i = blockIdx.x * (long long)blockDim.x + threadIdx.x; i < n;
         i += (long long)gridDim.x * blockDim.x)
        out[i] = (_Float16)in[i];
}

__global__ void att_scale_cq(const float* __restrict__ c, const float* __restrict__ cq,
                             _Float16* __restrict__ out, long long n, int H)
{
    for (long long i = blockIdx.x * (long long)blockDim.x + threadIdx.x; i < n;
         i += (long long)gridDim.x * blockDim.x)
        out[i] = (_Float16)(c[i] * cq[i % H]);
}

// s[b,c,q] += s0[b,c] + s1[b,q] + bias
__global__ void att_add_s(float* __restrict__ s, const float* __restrict__ s0,
                          const float* __restrict__ s1, const float* __restrict__ bias,
                          int B, int CL, int QL)
{
    long long n = (long long)B * CL * QL;
    for (long long i = blockIdx.x * (long long)blockDim.x + threadIdx.x; i < n;
         i += (long long)gridDim.x * blockDim.x) {
        int qq = (int)(i % QL);
        long long t = i / QL;
        int cc = (int)(t % CL);
        int b  = (int)(t / CL);
        s[i] += s0[b * CL + cc] + s1[b * QL + qq] + bias[0];
    }
}

// x = concat([c, a, c*a, c*b2, s3, ac], axis=-1) row-wise
__global__ void att_build_x(const float* __restrict__ c, const float* __restrict__ a,
                            const float* __restrict__ b2, const float* __restrict__ s3,
                            const float* __restrict__ ac, float* __restrict__ x,
                            long long total, int H, int E2)
{
    for (long long i = blockIdx.x * (long long)blockDim.x + threadIdx.x; i < total;
         i += (long long)gridDim.x * blockDim.x) {
        long long row = i / H;
        int h = (int)(i % H);
        float cv = c[i], av = a[i];
        long long base = row * (long long)E2;
        x[base + h]         = cv;
        x[base + H + h]     = av;
        x[base + 2 * H + h] = cv * av;
        x[base + 3 * H + h] = cv * b2[i];
        x[base + 4 * H + h] = s3[i];
        x[base + 5 * H + h] = ac[i];
    }
}

// ---------------------------------------------------------------------------
// Row dot:  out[row] = dot(x[row,:], w)   (one wave per row)
// ---------------------------------------------------------------------------
__global__ __launch_bounds__(128) void att_rowdot(const float* __restrict__ x,
                                                  const float* __restrict__ w,
                                                  float* __restrict__ out,
                                                  long long nRows, int H)
{
    const int wave = threadIdx.x >> 5, lane = threadIdx.x & 31;
    long long row = blockIdx.x * 4LL + wave;
    if (row >= nRows) return;
    const float* p = x + row * H;
    float s = 0.0f;
    for (int j = lane; j < H; j += 32) s += p[j] * w[j];
#pragma unroll
    for (int o = 16; o; o >>= 1) s += __shfl_xor(s, o, 32);
    if (lane == 0) out[row] = s;
}

// ---------------------------------------------------------------------------
// Masked softmax along last dim (one wave per row).  out = accW*softmax(scale*in [masked])
// optionally accumulated onto existing outF (accFirst=0).  C multiple of 32, C<=512.
// ---------------------------------------------------------------------------
__global__ __launch_bounds__(128) void att_softmax_rows(
    const float* __restrict__ in, const int* __restrict__ mask,
    float* __restrict__ outF, _Float16* __restrict__ outH,
    int R, int C, long long sIn, long long sOut, int maskStride,
    float scale, float accW, int accFirst, long long Rtotal)
{
    const int wave = threadIdx.x >> 5, lane = threadIdx.x & 31;
    long long row = blockIdx.x * 4LL + wave;
    if (row >= Rtotal) return;
    int b  = (int)(row / R);
    int rr = (int)(row % R);
    const float* p = in + (long long)b * sIn + (long long)rr * C;
    const int iters = C >> 5;
    float vals[16];
    float mx = -3.0e38f;
    for (int i = 0; i < iters; ++i) {
        int j = (i << 5) + lane;
        float v = p[j] * scale;
        if (mask && mask[b * maskStride + j] == 0) v = NEGBIG;
        vals[i] = v;
        mx = fmaxf(mx, v);
    }
#pragma unroll
    for (int o = 16; o; o >>= 1) mx = fmaxf(mx, __shfl_xor(mx, o, 32));
    float sum = 0.0f;
    for (int i = 0; i < iters; ++i) { vals[i] = expf(vals[i] - mx); sum += vals[i]; }
#pragma unroll
    for (int o = 16; o; o >>= 1) sum += __shfl_xor(sum, o, 32);
    const float inv = 1.0f / sum;
    float* po = outF + (long long)b * sOut + (long long)rr * C;
    _Float16* ph = outH ? outH + (long long)b * sOut + (long long)rr * C : (_Float16*)0;
    for (int i = 0; i < iters; ++i) {
        int j = (i << 5) + lane;
        float v = vals[i] * inv * accW;
        if (!accFirst) v += po[j];
        po[j] = v;
        if (ph) ph[j] = (_Float16)v;
    }
}

// ---------------------------------------------------------------------------
// Masked softmax along axis-1 (length R over stride C); one wave per (b,col).
// R multiple of 32, R<=512.  mask indexed by the softmax-axis position.
// ---------------------------------------------------------------------------
__global__ __launch_bounds__(128) void att_softmax_cols(
    const float* __restrict__ in, const int* __restrict__ mask,
    float* __restrict__ outF, _Float16* __restrict__ outH,
    int R, int C, long long sIn, long long sOut, int maskStride,
    float scale, long long Ctotal)
{
    const int wave = threadIdx.x >> 5, lane = threadIdx.x & 31;
    long long col = blockIdx.x * 4LL + wave;
    if (col >= Ctotal) return;
    int b  = (int)(col / C);
    int cc = (int)(col % C);
    const float* p = in + (long long)b * sIn + cc;
    const int iters = R >> 5;
    float vals[16];
    float mx = -3.0e38f;
    for (int i = 0; i < iters; ++i) {
        int rIdx = (i << 5) + lane;
        float v = p[(long long)rIdx * C] * scale;
        if (mask && mask[b * maskStride + rIdx] == 0) v = NEGBIG;
        vals[i] = v;
        mx = fmaxf(mx, v);
    }
#pragma unroll
    for (int o = 16; o; o >>= 1) mx = fmaxf(mx, __shfl_xor(mx, o, 32));
    float sum = 0.0f;
    for (int i = 0; i < iters; ++i) { vals[i] = expf(vals[i] - mx); sum += vals[i]; }
#pragma unroll
    for (int o = 16; o; o >>= 1) sum += __shfl_xor(sum, o, 32);
    const float inv = 1.0f / sum;
    float* po = outF + (long long)b * sOut + cc;
    _Float16* ph = outH ? outH + (long long)b * sOut + cc : (_Float16*)0;
    for (int i = 0; i < iters; ++i) {
        int rIdx = (i << 5) + lane;
        float v = vals[i] * inv;
        po[(long long)rIdx * C] = v;
        if (ph) ph[(long long)rIdx * C] = (_Float16)v;
    }
}

// ---------------------------------------------------------------------------
// Tiled transpose (B, R, C) f32  ->  (B, C, R) f16
// ---------------------------------------------------------------------------
__global__ __launch_bounds__(256) void att_transpose_f32_f16(
    const float* __restrict__ in, _Float16* __restrict__ out,
    int R, int C, long long sIn, long long sOut)
{
    __shared__ float tile[32][33];
    const int b  = blockIdx.z;
    const int c0 = blockIdx.x << 5;
    const int r0 = blockIdx.y << 5;
    const float* pin = in + (long long)b * sIn;
    _Float16* pout   = out + (long long)b * sOut;
    for (int i = threadIdx.y; i < 32; i += 8) {
        int rr = r0 + i, cc = c0 + threadIdx.x;
        tile[i][threadIdx.x] = (rr < R && cc < C) ? pin[(long long)rr * C + cc] : 0.0f;
    }
    __syncthreads();
    for (int i = threadIdx.y; i < 32; i += 8) {
        int cc = c0 + i, rr = r0 + threadIdx.x;
        if (cc < C && rr < R)
            pout[(long long)cc * R + rr] = (_Float16)tile[threadIdx.x][i];
    }
}

// ---------------------------------------------------------------------------
// LayerNorm over last dim C (block of 256 per row); writes f32 and f16 copies.
// ---------------------------------------------------------------------------
__global__ __launch_bounds__(256) void att_layernorm(
    const float* __restrict__ x, const float* __restrict__ g, const float* __restrict__ be,
    float* __restrict__ y, _Float16* __restrict__ yh, int C)
{
    const long long row = blockIdx.x;
    const float* p = x + row * C;
    float s = 0.0f, s2 = 0.0f;
    for (int j = threadIdx.x; j < C; j += 256) { float v = p[j]; s += v; s2 += v * v; }
#pragma unroll
    for (int o = 16; o; o >>= 1) { s += __shfl_xor(s, o, 32); s2 += __shfl_xor(s2, o, 32); }
    __shared__ float w1[8], w2[8];
    const int wave = threadIdx.x >> 5, lane = threadIdx.x & 31;
    if (lane == 0) { w1[wave] = s; w2[wave] = s2; }
    __syncthreads();
    s = 0.0f; s2 = 0.0f;
#pragma unroll
    for (int w = 0; w < 8; ++w) { s += w1[w]; s2 += w2[w]; }
    const float mean = s / C;
    const float var  = s2 / C - mean * mean;
    const float rstd = rsqrtf(var + 1e-5f);
    for (int j = threadIdx.x; j < C; j += 256) {
        float v = (p[j] - mean) * rstd * g[j] + be[j];
        y[row * C + j]  = v;
        yh[row * C + j] = (_Float16)v;
    }
}

// ---------------------------------------------------------------------------
// Host-side launch helpers
// ---------------------------------------------------------------------------
static inline void gemm(hipStream_t st, const _Float16* A, const _Float16* W,
                        const float* bias, const float* Cadd, float* outF, _Float16* outH,
                        int Bn, int M, int N, int K, int lda, int ldw, int ldo,
                        long long sA, long long sW, long long sO, long long sC, float alpha)
{
    dim3 grd((N + 255) / 256, (M + 31) / 32, Bn);
    att_gemm_f16_wmma<<<grd, 128, 0, st>>>(A, W, bias, Cadd, outF, outH,
                                           M, N, K, lda, ldw, ldo, sA, sW, sO, sC, alpha);
}

static inline void transp(hipStream_t st, const float* in, _Float16* out,
                          int Bn, int R, int C, long long sIn, long long sOut)
{
    dim3 grd(C / 32, R / 32, Bn);
    att_transpose_f32_f16<<<grd, dim3(32, 8, 1), 0, st>>>(in, out, R, C, sIn, sOut);
}

extern "C" void kernel_launch(void* const* d_in, const int* in_sizes, int n_in,
                              void* d_out, int out_size, void* d_ws, size_t ws_size,
                              hipStream_t stream)
{
    (void)in_sizes; (void)n_in; (void)out_size; (void)ws_size;

    const int B = 8, CL = 512, QL = 64, H = 768, E2 = 4608, HD = 192;
    const float isq = 0.07216878364870323f;  // 1/sqrt(192)

    const float* c        = (const float*)d_in[0];
    const float* q        = (const float*)d_in[1];
    const int*   c_mask   = (const int*)d_in[2];
    const int*   q_mask   = (const int*)d_in[3];
    const float* c_weight = (const float*)d_in[4];
    const float* q_weight = (const float*)d_in[5];
    const float* cq_w     = (const float*)d_in[6];
    const float* bias0    = (const float*)d_in[7];
    const float* wq1      = (const float*)d_in[8];
    const float* bq1      = (const float*)d_in[9];
    const float* wk1      = (const float*)d_in[10];
    const float* bk1      = (const float*)d_in[11];
    const float* gamma    = (const float*)d_in[12];
    const float* beta     = (const float*)d_in[13];
    const float* wq2      = (const float*)d_in[14];
    const float* bq2      = (const float*)d_in[15];
    const float* wk2      = (const float*)d_in[16];
    const float* bk2      = (const float*)d_in[17];

    // ---- workspace bump allocator (deterministic, recomputed every call) ----
    char* ws = (char*)d_ws;
    size_t off = 0;
    auto alloc = [&](size_t bytes) -> void* {
        off = (off + 255) & ~(size_t)255;
        void* p = ws + off;
        off += bytes;
        return p;
    };
    auto fbuf = [&](long long n) { return (float*)alloc((size_t)n * 4); };
    auto hbuf = [&](long long n) { return (_Float16*)alloc((size_t)n * 2); };

    _Float16* wq1_h = hbuf((long long)H * H);
    _Float16* wk1_h = hbuf((long long)H * H);
    _Float16* wq2_h = hbuf((long long)E2 * E2);
    _Float16* wk2_h = hbuf((long long)E2 * E2);
    _Float16* c_h   = hbuf((long long)B * CL * H);
    _Float16* q_h   = hbuf((long long)B * QL * H);
    _Float16* cs_h  = hbuf((long long)B * CL * H);
    _Float16* qT_h  = hbuf((long long)B * H * QL);
    _Float16* cT_h  = hbuf((long long)B * H * CL);
    float*    s0    = fbuf((long long)B * CL);
    float*    s1q   = fbuf((long long)B * QL);
    float*    sbuf  = fbuf((long long)B * CL * QL);   // s, later per-head score tmp
    float*    s1mF  = fbuf((long long)B * CL * QL);
    _Float16* s1m_h = hbuf((long long)B * CL * QL);
    float*    s2mF  = fbuf((long long)B * CL * QL);
    _Float16* s2m_h = hbuf((long long)B * CL * QL);
    float*    aF    = fbuf((long long)B * CL * H);
    _Float16* t_h   = hbuf((long long)B * CL * CL);
    float*    bF    = fbuf((long long)B * CL * H);
    _Float16* qh_h  = hbuf((long long)B * CL * H);
    _Float16* kh_h  = hbuf((long long)B * QL * H);
    float*    scoatF   = fbuf((long long)B * CL * QL);
    float*    scoat1F  = fbuf((long long)B * CL * QL);
    _Float16* scoat1_h = hbuf((long long)B * CL * QL);
    float*    scoat2F  = fbuf((long long)B * CL * QL);
    _Float16* sc2T_h   = hbuf((long long)B * QL * CL);
    float*    acoatF   = fbuf((long long)B * CL * H);
    float*    bcoatF   = fbuf((long long)B * QL * H);
    _Float16* bcoatT_h = hbuf((long long)B * H * QL);
    float*    scoat3F  = fbuf((long long)B * CL * H);
    float*    xF    = fbuf((long long)B * CL * E2);
    float*    yF    = fbuf((long long)B * CL * E2);
    _Float16* y_h   = hbuf((long long)B * CL * E2);
    _Float16* yT_h  = hbuf((long long)B * E2 * CL);
    _Float16* qh2_h = hbuf((long long)B * CL * E2);
    _Float16* kh2_h = hbuf((long long)B * CL * E2);
    float*    sstmp = fbuf((long long)B * CL * CL);
    float*    ssF   = fbuf((long long)B * CL * CL);
    _Float16* ss1_h = hbuf((long long)B * CL * CL);

    auto cvt = [&](const float* in, _Float16* out, long long n) {
        att_f32_to_f16<<<dim3((unsigned)((n + 255) / 256)), 256, 0, stream>>>(in, out, n);
    };

    // ---- 1. precision conversion of weights / activations ----
    cvt(wq1, wq1_h, (long long)H * H);
    cvt(wk1, wk1_h, (long long)H * H);
    cvt(wq2, wq2_h, (long long)E2 * E2);
    cvt(wk2, wk2_h, (long long)E2 * E2);
    cvt(c, c_h, (long long)B * CL * H);
    cvt(q, q_h, (long long)B * QL * H);
    {
        long long n = (long long)B * CL * H;
        att_scale_cq<<<dim3((unsigned)((n + 255) / 256)), 256, 0, stream>>>(c, cq_w, cs_h, n, H);
    }
    transp(stream, q, qT_h, B, QL, H, (long long)QL * H, (long long)H * QL);
    transp(stream, c, cT_h, B, CL, H, (long long)CL * H, (long long)H * CL);

    // ---- 2. trilinear scores s = c@cw + (q@qw)^T + (c*cq)@q^T + bias ----
    att_rowdot<<<dim3((B * CL + 3) / 4), 128, 0, stream>>>(c, c_weight, s0, (long long)B * CL, H);
    att_rowdot<<<dim3((B * QL + 3) / 4), 128, 0, stream>>>(q, q_weight, s1q, (long long)B * QL, H);
    gemm(stream, cs_h, q_h, nullptr, nullptr, sbuf, nullptr, B, CL, QL, H,
         H, H, QL, (long long)CL * H, (long long)QL * H, (long long)CL * QL, 0, 1.0f);
    {
        long long n = (long long)B * CL * QL;
        att_add_s<<<dim3((unsigned)((n + 255) / 256)), 256, 0, stream>>>(sbuf, s0, s1q, bias0, B, CL, QL);
    }

    // ---- 3. masked softmaxes of s ----
    att_softmax_rows<<<dim3((B * CL + 3) / 4), 128, 0, stream>>>(
        sbuf, q_mask, s1mF, s1m_h, CL, QL, (long long)CL * QL, (long long)CL * QL, QL,
        1.0f, 1.0f, 1, (long long)B * CL);
    att_softmax_cols<<<dim3((B * QL + 3) / 4), 128, 0, stream>>>(
        sbuf, c_mask, s2mF, s2m_h, CL, QL, (long long)CL * QL, (long long)CL * QL, CL,
        1.0f, (long long)B * QL);

    // ---- 4. a = s1m @ q ; t = s1m @ s2m^T ; b = t @ c ----
    gemm(stream, s1m_h, qT_h, nullptr, nullptr, aF, nullptr, B, CL, H, QL,
         QL, QL, H, (long long)CL * QL, (long long)H * QL, (long long)CL * H, 0, 1.0f);
    gemm(stream, s1m_h, s2m_h, nullptr, nullptr, nullptr, t_h, B, CL, CL, QL,
         QL, QL, CL, (long long)CL * QL, (long long)CL * QL, (long long)CL * CL, 0, 1.0f);
    gemm(stream, t_h, cT_h, nullptr, nullptr, bF, nullptr, B, CL, H, CL,
         CL, CL, H, (long long)CL * CL, (long long)H * CL, (long long)CL * H, 0, 1.0f);

    // ---- 5. mha_probs(c, q, wq1, wk1, 4 heads) -> scoat ----
    gemm(stream, c_h, wq1_h, bq1, nullptr, nullptr, qh_h, B, CL, H, H,
         H, H, H, (long long)CL * H, 0, (long long)CL * H, 0, 1.0f);
    gemm(stream, q_h, wk1_h, bk1, nullptr, nullptr, kh_h, B, QL, H, H,
         H, H, H, (long long)QL * H, 0, (long long)QL * H, 0, 1.0f);
    for (int h = 0; h < 4; ++h) {
        gemm(stream, qh_h + (long long)h * HD, kh_h + (long long)h * HD, nullptr, nullptr,
             sbuf, nullptr, B, CL, QL, HD, H, H, QL,
             (long long)CL * H, (long long)QL * H, (long long)CL * QL, 0, isq);
        att_softmax_rows<<<dim3((B * CL + 3) / 4), 128, 0, stream>>>(
            sbuf, nullptr, scoatF, nullptr, CL, QL, (long long)CL * QL, (long long)CL * QL, QL,
            1.0f, 0.25f, (h == 0) ? 1 : 0, (long long)B * CL);
    }

    // ---- 6. scoat1 / scoat2, acoat, bcoat, scoat3 ----
    att_softmax_rows<<<dim3((B * CL + 3) / 4), 128, 0, stream>>>(
        scoatF, q_mask, scoat1F, scoat1_h, CL, QL, (long long)CL * QL, (long long)CL * QL, QL,
        1.0f, 1.0f, 1, (long long)B * CL);
    att_softmax_cols<<<dim3((B * QL + 3) / 4), 128, 0, stream>>>(
        scoatF, c_mask, scoat2F, nullptr, CL, QL, (long long)CL * QL, (long long)CL * QL, CL,
        1.0f, (long long)B * QL);
    transp(stream, scoat2F, sc2T_h, B, CL, QL, (long long)CL * QL, (long long)QL * CL);
    gemm(stream, scoat1_h, qT_h, nullptr, nullptr, acoatF, nullptr, B, CL, H, QL,
         QL, QL, H, (long long)CL * QL, (long long)H * QL, (long long)CL * H, 0, 1.0f);
    gemm(stream, sc2T_h, cT_h, nullptr, nullptr, bcoatF, nullptr, B, QL, H, CL,
         CL, CL, H, (long long)QL * CL, (long long)H * CL, (long long)QL * H, 0, 1.0f);
    transp(stream, bcoatF, bcoatT_h, B, QL, H, (long long)QL * H, (long long)H * QL);
    gemm(stream, scoat1_h, bcoatT_h, nullptr, nullptr, scoat3F, nullptr, B, CL, H, QL,
         QL, QL, H, (long long)CL * QL, (long long)H * QL, (long long)CL * H, 0, 1.0f);

    // ---- 7. x = concat(...), y = layernorm(x) ----
    {
        long long n = (long long)B * CL * H;
        att_build_x<<<dim3((unsigned)((n + 255) / 256)), 256, 0, stream>>>(
            c, aF, bF, scoat3F, acoatF, xF, n, H, E2);
    }
    att_layernorm<<<dim3(B * CL), 256, 0, stream>>>(xF, gamma, beta, yF, y_h, E2);
    transp(stream, yF, yT_h, B, CL, E2, (long long)CL * E2, (long long)E2 * CL);

    // ---- 8. mha_probs(y, y, wq2, wk2, 24 heads) -> ss (per-head accumulate) ----
    gemm(stream, y_h, wq2_h, bq2, nullptr, nullptr, qh2_h, B, CL, E2, E2,
         E2, E2, E2, (long long)CL * E2, 0, (long long)CL * E2, 0, 1.0f);
    gemm(stream, y_h, wk2_h, bk2, nullptr, nullptr, kh2_h, B, CL, E2, E2,
         E2, E2, E2, (long long)CL * E2, 0, (long long)CL * E2, 0, 1.0f);
    for (int h = 0; h < 24; ++h) {
        gemm(stream, qh2_h + (long long)h * HD, kh2_h + (long long)h * HD, nullptr, nullptr,
             sstmp, nullptr, B, CL, CL, HD, E2, E2, CL,
             (long long)CL * E2, (long long)CL * E2, (long long)CL * CL, 0, isq);
        att_softmax_rows<<<dim3((B * CL + 3) / 4), 128, 0, stream>>>(
            sstmp, nullptr, ssF, nullptr, CL, CL, (long long)CL * CL, (long long)CL * CL, CL,
            1.0f, 1.0f / 24.0f, (h == 0) ? 1 : 0, (long long)B * CL);
    }

    // ---- 9. ss1 = masked softmax over axis 1; patt = ss1 @ y + x ----
    att_softmax_cols<<<dim3((B * CL + 3) / 4), 128, 0, stream>>>(
        ssF, c_mask, ssF, ss1_h, CL, CL, (long long)CL * CL, (long long)CL * CL, CL,
        1.0f, (long long)B * CL);
    gemm(stream, ss1_h, yT_h, nullptr, xF, (float*)d_out, nullptr, B, CL, E2, CL,
         CL, CL, E2, (long long)CL * CL, (long long)E2 * CL, (long long)CL * E2,
         (long long)CL * E2, 1.0f);
}
